// Block_26001732010173
// MI455X (gfx1250) — matrix operand order repaired and checked by reading, not compiled, not run on hardware
//
#include <hip/hip_runtime.h>
#include <hip/hip_bf16.h>

typedef __bf16 bf16;
typedef bf16  v16bf __attribute__((ext_vector_type(16)));
typedef bf16  v8bf  __attribute__((ext_vector_type(8)));
typedef float v8f   __attribute__((ext_vector_type(8)));

#define C_DIM 768
#define NHEAD 12
#define HDIM  64
#define B_SZ  8
#define T_SZ  16
#define K_SZ  196
#define KT    (K_SZ * T_SZ)        // 3136
#define M_TOK (KT + 1)             // 3137
#define N_SP  (K_SZ + 1)           // 197

union FragBF { v16bf v; v8bf h[2]; };

// ---------------------------------------------------------------------------
// Generic elementwise f32 -> bf16 convert
// ---------------------------------------------------------------------------
__global__ __launch_bounds__(256) void cvt_f32_bf16(const float* __restrict__ s,
                                                    bf16* __restrict__ d, size_t n) {
  size_t i = (size_t)blockIdx.x * 256 + threadIdx.x;
  if (i < n) d[i] = (bf16)s[i];
}

// ---------------------------------------------------------------------------
// Block reduction helper (256 threads)
// ---------------------------------------------------------------------------
__device__ __forceinline__ float block_reduce(float v, float* red) {
  int tid = threadIdx.x;
  red[tid] = v;
  __syncthreads();
  for (int s = 128; s > 0; s >>= 1) {
    if (tid < s) red[tid] += red[tid + s];
    __syncthreads();
  }
  float r = red[0];
  __syncthreads();
  return r;
}

// ---------------------------------------------------------------------------
// LayerNorm variants (one block = one row of 768)
// ---------------------------------------------------------------------------
template <typename OutT>
__global__ __launch_bounds__(256) void ln_rows(const float* __restrict__ src, int rowStride,
                                               const float* __restrict__ w,
                                               const float* __restrict__ bia,
                                               OutT* __restrict__ out) {
  __shared__ float red[256];
  int r = blockIdx.x, tid = threadIdx.x;
  const float* s = src + (size_t)r * rowStride * C_DIM;
  float v0 = s[tid], v1 = s[tid + 256], v2 = s[tid + 512];
  float mean = block_reduce(v0 + v1 + v2, red) * (1.0f / C_DIM);
  float d0 = v0 - mean, d1 = v1 - mean, d2 = v2 - mean;
  float var = block_reduce(d0 * d0 + d1 * d1 + d2 * d2, red) * (1.0f / C_DIM);
  float rstd = rsqrtf(var + 1e-5f);
  OutT* o = out + (size_t)r * C_DIM;
  o[tid]       = (OutT)(d0 * rstd * w[tid]       + bia[tid]);
  o[tid + 256] = (OutT)(d1 * rstd * w[tid + 256] + bia[tid + 256]);
  o[tid + 512] = (OutT)(d2 * rstd * w[tid + 512] + bia[tid + 512]);
}

// xt = x[:,1:,:] reshaped to (B*K, T, C); row r -> x row r + r/3136 + 1
__global__ __launch_bounds__(256) void ln_xt(const float* __restrict__ x,
                                             const float* __restrict__ w,
                                             const float* __restrict__ bia,
                                             bf16* __restrict__ out) {
  __shared__ float red[256];
  int r = blockIdx.x, tid = threadIdx.x;
  size_t srcRow = (size_t)r + (size_t)(r / KT) + 1;
  const float* s = x + srcRow * C_DIM;
  float v0 = s[tid], v1 = s[tid + 256], v2 = s[tid + 512];
  float mean = block_reduce(v0 + v1 + v2, red) * (1.0f / C_DIM);
  float d0 = v0 - mean, d1 = v1 - mean, d2 = v2 - mean;
  float var = block_reduce(d0 * d0 + d1 * d1 + d2 * d2, red) * (1.0f / C_DIM);
  float rstd = rsqrtf(var + 1e-5f);
  bf16* o = out + (size_t)r * C_DIM;
  o[tid]       = (bf16)(d0 * rstd * w[tid]       + bia[tid]);
  o[tid + 256] = (bf16)(d1 * rstd * w[tid + 256] + bia[tid + 256]);
  o[tid + 512] = (bf16)(d2 * rstd * w[tid + 512] + bia[tid + 512]);
}

// xs: row r over (B*T, 197): n==0 -> x[b,0,:], else xt_full[b, (n-1)*T + t]
__global__ __launch_bounds__(256) void ln_xs(const float* __restrict__ x,
                                             const float* __restrict__ xtf,
                                             const float* __restrict__ w,
                                             const float* __restrict__ bia,
                                             bf16* __restrict__ out) {
  __shared__ float red[256];
  int r = blockIdx.x, tid = threadIdx.x;
  int bt = r / N_SP, n = r % N_SP;
  int b = bt / T_SZ, t = bt % T_SZ;
  const float* s;
  if (n == 0) s = x + (size_t)b * M_TOK * C_DIM;
  else        s = xtf + ((size_t)b * KT + (size_t)(n - 1) * T_SZ + t) * C_DIM;
  float v0 = s[tid], v1 = s[tid + 256], v2 = s[tid + 512];
  float mean = block_reduce(v0 + v1 + v2, red) * (1.0f / C_DIM);
  float d0 = v0 - mean, d1 = v1 - mean, d2 = v2 - mean;
  float var = block_reduce(d0 * d0 + d1 * d1 + d2 * d2, red) * (1.0f / C_DIM);
  float rstd = rsqrtf(var + 1e-5f);
  bf16* o = out + (size_t)r * C_DIM;
  o[tid]       = (bf16)(d0 * rstd * w[tid]       + bia[tid]);
  o[tid + 256] = (bf16)(d1 * rstd * w[tid + 256] + bia[tid + 256]);
  o[tid + 512] = (bf16)(d2 * rstd * w[tid + 512] + bia[tid + 512]);
}

// ---------------------------------------------------------------------------
// WMMA bf16 GEMM: C[M,N] = A[M,Kd](bf16) * Bw[Kd,N](bf16) + bias (+ residual)
// One wave -> 32x32 output (2x2 WMMA tiles, 4 f32 accumulators); block = 8
// waves as 2x4 wave-tiles -> 64x128 block tile.  Per K-step of 32 each lane
// issues 4 global_load_b128 (A) + 32 scalar bf16 loads (B) feeding 4 WMMAs (32B/lane/WMMA operand traffic).
// EPI: 0 none, 1 exact GELU.  RESMODE: 0 none, 1 direct (resid[row*ldc+col]),
// 2 temporal map (resid row = row + row/resParam + 1, stride N).
// asub > 0: A row = (row/asub)*16 + (16-asub) + row%asub  (h[:, T-asub:] slice)
// ---------------------------------------------------------------------------
template <int EPI, int RESMODE, typename OutT>
__global__ __launch_bounds__(256)
void gemm_bf16_wmma(const bf16* __restrict__ A, int lda, int asub,
                    const bf16* __restrict__ Bw, int ldb,
                    OutT* __restrict__ Cout, int ldc,
                    int M, int N, int Kd,
                    const float* __restrict__ bias,
                    const float* __restrict__ resid, int resParam) {
  const int lane = threadIdx.x & 31;
  const int wave = threadIdx.x >> 5;
  const int wy = wave >> 2, wx = wave & 3;
  const int rowBase = blockIdx.y * 64 + wy * 32;
  const int colBase = blockIdx.x * 128 + wx * 32;
  if (colBase >= N) return;

  const int m = lane & 15;
  const int half = lane >> 4;

  auto mapRow = [&](int r) -> size_t {
    int lr = r < M ? r : (M - 1);
    if (asub > 0) lr = (lr / asub) * 16 + (16 - asub) + (lr % asub);
    return (size_t)lr;
  };
  const bf16* Ap0 = A + mapRow(rowBase + m)      * lda + 8 * half;
  const bf16* Ap1 = A + mapRow(rowBase + 16 + m) * lda + 8 * half;
  const bf16* Bp  = Bw + (size_t)(8 * half) * ldb + colBase + m;

  v8f acc00 = {}, acc01 = {}, acc10 = {}, acc11 = {};
  for (int k0 = 0; k0 < Kd; k0 += 32) {
    FragBF a0, a1, b0, b1;
    a0.h[0] = *(const v8bf*)(Ap0 + k0);
    a0.h[1] = *(const v8bf*)(Ap0 + k0 + 16);
    a1.h[0] = *(const v8bf*)(Ap1 + k0);
    a1.h[1] = *(const v8bf*)(Ap1 + k0 + 16);
    const bf16* brow = Bp + (size_t)k0 * ldb;
    for (int s = 0; s < 8; ++s) {
      b0.h[0][s] = brow[(size_t)s * ldb];
      b0.h[1][s] = brow[(size_t)(16 + s) * ldb];
      b1.h[0][s] = brow[(size_t)s * ldb + 16];
      b1.h[1][s] = brow[(size_t)(16 + s) * ldb + 16];
    }
    if (k0 + 32 < Kd) {
      __builtin_prefetch(Ap0 + k0 + 32, 0, 0);
      __builtin_prefetch(Ap1 + k0 + 32, 0, 0);
      __builtin_prefetch(brow + (size_t)32 * ldb, 0, 0);
    }
    acc00 = __builtin_amdgcn_wmma_f32_16x16x32_bf16(false, a0.v, false, b0.v,
                                                    (short)0, acc00, false, false);
    acc01 = __builtin_amdgcn_wmma_f32_16x16x32_bf16(false, a0.v, false, b1.v,
                                                    (short)0, acc01, false, false);
    acc10 = __builtin_amdgcn_wmma_f32_16x16x32_bf16(false, a1.v, false, b0.v,
                                                    (short)0, acc10, false, false);
    acc11 = __builtin_amdgcn_wmma_f32_16x16x32_bf16(false, a1.v, false, b1.v,
                                                    (short)0, acc11, false, false);
  }

  auto store_tile = [&](v8f& acc, int ro, int co) {
    const int col = colBase + co + m;
    float bv = bias ? bias[col] : 0.0f;
#pragma unroll
    for (int r = 0; r < 8; ++r) {
      int orow = rowBase + ro + r + 8 * half;
      if (orow >= M) continue;
      float v = acc[r] + bv;
      if (EPI == 1) v = 0.5f * v * (1.0f + erff(v * 0.70710678118f));
      if (RESMODE == 1) v += resid[(size_t)orow * ldc + col];
      else if (RESMODE == 2) {
        int rr = orow + orow / resParam + 1;
        v += resid[(size_t)rr * N + col];
      }
      Cout[(size_t)orow * ldc + col] = (OutT)v;
    }
  };
  store_tile(acc00, 0, 0);
  store_tile(acc01, 0, 16);
  store_tile(acc10, 16, 0);
  store_tile(acc11, 16, 16);
}

// ---------------------------------------------------------------------------
// Temporal attention: one block per (b*K) group, seq len t <= 16, 12 heads.
// qkv layout per row: [q(768) | k(768) | v(768)], head h cols h*64..h*64+63.
// ---------------------------------------------------------------------------
__global__ __launch_bounds__(256) void temporal_attn(const float* __restrict__ qkv,
                                                     bf16* __restrict__ ctx, int t) {
  __shared__ float qs[16 * 64], ks[16 * 64], vs[16 * 64];
  __shared__ float sb[16 * 17], mx[16], dn[16];
  int bk = blockIdx.x, tid = threadIdx.x;
  for (int h = 0; h < NHEAD; ++h) {
    for (int i = tid; i < t * 64; i += 256) {
      int ti = i >> 6, d = i & 63;
      size_t base = ((size_t)bk * t + ti) * 2304 + h * 64 + d;
      qs[i] = qkv[base];
      ks[i] = qkv[base + 768];
      vs[i] = qkv[base + 1536];
    }
    __syncthreads();
    for (int i = tid; i < t * t; i += 256) {
      int qi = i / t, kj = i % t;
      float s = 0.f;
      for (int d = 0; d < 64; ++d) s += qs[qi * 64 + d] * ks[kj * 64 + d];
      sb[qi * 17 + kj] = s * 0.125f;  // 1/sqrt(64)
    }
    __syncthreads();
    if (tid < t) {
      float mm = -1e30f;
      for (int j = 0; j < t; ++j) mm = fmaxf(mm, sb[tid * 17 + j]);
      float den = 0.f;
      for (int j = 0; j < t; ++j) den += __expf(sb[tid * 17 + j] - mm);
      mx[tid] = mm; dn[tid] = den;
    }
    __syncthreads();
    for (int i = tid; i < t * t; i += 256) {
      int qi = i / t, kj = i % t;
      sb[qi * 17 + kj] = __expf(sb[qi * 17 + kj] - mx[qi]) / dn[qi];
    }
    __syncthreads();
    for (int i = tid; i < t * 64; i += 256) {
      int qi = i >> 6, d = i & 63;
      float a = 0.f;
      for (int j = 0; j < t; ++j) a += sb[qi * 17 + j] * vs[j * 64 + d];
      ctx[((size_t)bk * t + qi) * C_DIM + h * 64 + d] = (bf16)a;
    }
    __syncthreads();
  }
}

// ---------------------------------------------------------------------------
// Spatial attention: grid (12 heads, 128 batch*time), seq 197, LDS-staged K/V.
// ---------------------------------------------------------------------------
__global__ __launch_bounds__(256) void spatial_attn(const float* __restrict__ qkv,
                                                    bf16* __restrict__ ctx) {
  __shared__ bf16 kb[N_SP * 64], vb[N_SP * 64];
  __shared__ float qsh[8 * 64], sb[8 * 200], mx[8], dn[8];
  int h = blockIdx.x, bt = blockIdx.y, tid = threadIdx.x;
  for (int i = tid; i < N_SP * 64; i += 256) {
    int n = i >> 6, d = i & 63;
    size_t base = ((size_t)bt * N_SP + n) * 2304 + h * 64 + d;
    kb[i] = (bf16)qkv[base + 768];
    vb[i] = (bf16)qkv[base + 1536];
  }
  __syncthreads();
  for (int q0 = 0; q0 < N_SP; q0 += 8) {
    int nq = min(8, N_SP - q0);
    for (int i = tid; i < nq * 64; i += 256) {
      int qi = i >> 6, d = i & 63;
      qsh[i] = qkv[((size_t)bt * N_SP + q0 + qi) * 2304 + h * 64 + d];
    }
    __syncthreads();
    for (int i = tid; i < nq * N_SP; i += 256) {
      int qi = i / N_SP, j = i % N_SP;
      float s = 0.f;
      for (int d = 0; d < 64; ++d) s += qsh[qi * 64 + d] * (float)kb[j * 64 + d];
      sb[qi * 200 + j] = s * 0.125f;
    }
    __syncthreads();
    if (tid < nq) {
      float mm = -1e30f;
      for (int j = 0; j < N_SP; ++j) mm = fmaxf(mm, sb[tid * 200 + j]);
      float den = 0.f;
      for (int j = 0; j < N_SP; ++j) den += __expf(sb[tid * 200 + j] - mm);
      mx[tid] = mm; dn[tid] = den;
    }
    __syncthreads();
    for (int i = tid; i < nq * N_SP; i += 256) {
      int qi = i / N_SP, j = i % N_SP;
      sb[qi * 200 + j] = __expf(sb[qi * 200 + j] - mx[qi]) / dn[qi];
    }
    __syncthreads();
    for (int i = tid; i < nq * 64; i += 256) {
      int qi = i >> 6, d = i & 63;
      float a = 0.f;
      for (int j = 0; j < N_SP; ++j) a += sb[qi * 200 + j] * (float)vb[j * 64 + d];
      ctx[((size_t)bt * N_SP + q0 + qi) * C_DIM + h * 64 + d] = (bf16)a;
    }
    __syncthreads();
  }
}

// ---------------------------------------------------------------------------
// Scale-merge: dst[:, tBig/2:] = 0.5*dst + 0.5*srcSmall
// ---------------------------------------------------------------------------
__global__ __launch_bounds__(256) void merge_half(float* __restrict__ dst,
                                                  const float* __restrict__ srcSmall,
                                                  int tBig, int nBK) {
  int tS = tBig >> 1;
  size_t idx = (size_t)blockIdx.x * 256 + threadIdx.x;
  size_t total = (size_t)nBK * tS * C_DIM;
  if (idx >= total) return;
  int c = (int)(idx % C_DIM);
  int r = (int)(idx / C_DIM);
  int bk = r / tS, ts = r % tS;
  size_t drow = (size_t)bk * tBig + tS + ts;
  dst[drow * C_DIM + c] = 0.5f * dst[drow * C_DIM + c] + 0.5f * srcSmall[(size_t)r * C_DIM + c];
}

// ---------------------------------------------------------------------------
// CLS aggregation: a = softmax(<cls[b,15], cls[b,t]>); out[b] = sum_t a_t cls[b,t]
// ---------------------------------------------------------------------------
__global__ __launch_bounds__(256) void cls_agg_kernel(const float* __restrict__ cls,
                                                      float* __restrict__ out) {
  __shared__ float wts[16];
  int b = blockIdx.x, tid = threadIdx.x;
  if (tid < 16) {
    const float* tgt = cls + ((size_t)b * 16 + 15) * C_DIM;
    const float* row = cls + ((size_t)b * 16 + tid) * C_DIM;
    float s = 0.f;
    for (int c = 0; c < C_DIM; ++c) s += tgt[c] * row[c];
    wts[tid] = s;
  }
  __syncthreads();
  if (tid == 0) {
    float mm = -1e30f;
    for (int t = 0; t < 16; ++t) mm = fmaxf(mm, wts[t]);
    float den = 0.f;
    for (int t = 0; t < 16; ++t) { wts[t] = __expf(wts[t] - mm); den += wts[t]; }
    for (int t = 0; t < 16; ++t) wts[t] /= den;
  }
  __syncthreads();
  for (int c = tid; c < C_DIM; c += 256) {
    float a = 0.f;
    for (int t = 0; t < 16; ++t) a += wts[t] * cls[((size_t)b * 16 + t) * C_DIM + c];
    out[(size_t)b * C_DIM + c] = a;
  }
}

// ---------------------------------------------------------------------------
// x_cat = concat(init_cls, xt_full) + concat(cls_agg, x_s)
// ---------------------------------------------------------------------------
__global__ __launch_bounds__(256) void build_xcat(const float* __restrict__ x,
                                                  const float* __restrict__ xtf,
                                                  const float* __restrict__ rsp,
                                                  const float* __restrict__ clsagg,
                                                  float* __restrict__ xcat) {
  size_t idx = (size_t)blockIdx.x * 256 + threadIdx.x;
  size_t total = (size_t)B_SZ * M_TOK * C_DIM;
  if (idx >= total) return;
  int c = (int)(idx % C_DIM);
  size_t row = idx / C_DIM;
  int b = (int)(row / M_TOK);
  int j = (int)(row % M_TOK);
  float v;
  if (j == 0) {
    v = x[idx] + clsagg[(size_t)b * C_DIM + c];
  } else {
    int jj = j - 1;
    int k = jj / T_SZ, t = jj % T_SZ;
    v = xtf[((size_t)b * KT + jj) * C_DIM + c] +
        rsp[(((size_t)b * T_SZ + t) * N_SP + 1 + k) * C_DIM + c];
  }
  xcat[idx] = v;
}

// ---------------------------------------------------------------------------
// Host launcher
// ---------------------------------------------------------------------------
extern "C" void kernel_launch(void* const* d_in, const int* in_sizes, int n_in,
                              void* d_out, int out_size, void* d_ws, size_t ws_size,
                              hipStream_t stream) {
  const float* x       = (const float*)d_in[0];
  const float* tn1_w   = (const float*)d_in[1];
  const float* tn1_b   = (const float*)d_in[2];
  const float* qkv4_w  = (const float*)d_in[3];
  const float* proj4_w = (const float*)d_in[4];
  const float* proj4_b = (const float*)d_in[5];
  const float* qkv8_w  = (const float*)d_in[6];
  const float* proj8_w = (const float*)d_in[7];
  const float* proj8_b = (const float*)d_in[8];
  const float* qkv16_w = (const float*)d_in[9];
  const float* proj16_w= (const float*)d_in[10];
  const float* proj16_b= (const float*)d_in[11];
  const float* tfc_w   = (const float*)d_in[12];
  const float* tfc_b   = (const float*)d_in[13];
  const float* n1_w    = (const float*)d_in[14];
  const float* n1_b    = (const float*)d_in[15];
  const float* sqkv_w  = (const float*)d_in[16];
  const float* sproj_w = (const float*)d_in[17];
  const float* sproj_b = (const float*)d_in[18];
  const float* ncls_w  = (const float*)d_in[19];
  const float* ncls_b  = (const float*)d_in[20];
  const float* n2_w    = (const float*)d_in[21];
  const float* n2_b    = (const float*)d_in[22];
  const float* fc1_w   = (const float*)d_in[23];
  const float* fc1_b   = (const float*)d_in[24];
  const float* fc2_w   = (const float*)d_in[25];
  const float* fc2_b   = (const float*)d_in[26];
  float* out = (float*)d_out;
  char* ws = (char*)d_ws;

  // ---- scratch arena ----
  size_t off = 0;
  auto arena = [&](size_t bytes) { size_t r = off; off += (bytes + 255) & ~(size_t)255; return r; };
  const size_t C = C_DIM, C3 = 3 * C_DIM, C4 = 4 * C_DIM;
  const int Mt = B_SZ * KT;          // 25088 temporal rows
  const int Ms = B_SZ * T_SZ * N_SP; // 25216 spatial rows
  const int Mc = B_SZ * M_TOK;       // 25096 rows with cls

  bf16* wqkv4  = (bf16*)(ws + arena(C * C3 * 2));
  bf16* wqkv8  = (bf16*)(ws + arena(C * C3 * 2));
  bf16* wqkv16 = (bf16*)(ws + arena(C * C3 * 2));
  bf16* wsqkv  = (bf16*)(ws + arena(C * C3 * 2));
  bf16* wproj4 = (bf16*)(ws + arena(C * C * 2));
  bf16* wproj8 = (bf16*)(ws + arena(C * C * 2));
  bf16* wproj16= (bf16*)(ws + arena(C * C * 2));
  bf16* wtfc   = (bf16*)(ws + arena(C * C * 2));
  bf16* wsproj = (bf16*)(ws + arena(C * C * 2));
  bf16* wfc1   = (bf16*)(ws + arena(C * C4 * 2));
  bf16* wfc2   = (bf16*)(ws + arena(C4 * C * 2));
  bf16* Abf    = (bf16*)(ws + arena((size_t)Ms * C * 2));   // LN outputs (h / xs_ln / h2)
  float* Qf    = (float*)(ws + arena((size_t)Ms * C3 * 4)); // qkv scratch, reused as act1(bf16)
  bf16* ctxbf  = (bf16*)(ws + arena((size_t)Ms * C * 2));   // attention context / res_temp bf16
  float* X4    = (float*)(ws + arena((size_t)(B_SZ * K_SZ * 4) * C * 4));
  float* X8    = (float*)(ws + arena((size_t)(B_SZ * K_SZ * 8) * C * 4));
  float* X16   = (float*)(ws + arena((size_t)Mt * C * 4));
  float* XTF   = (float*)(ws + arena((size_t)Mt * C * 4));
  float* RSP   = (float*)(ws + arena((size_t)Ms * C * 4));
  float* CLSsp = (float*)(ws + arena((size_t)128 * C * 4));
  float* CLSag = (float*)(ws + arena((size_t)B_SZ * C * 4));
  float* XCAT  = (float*)(ws + arena((size_t)Mc * C * 4));
  (void)ws_size; (void)in_sizes; (void)n_in; (void)out_size;

  // ---- weight conversion to bf16 ----
  struct CvtJob { const float* s; bf16* d; size_t n; } jobs[] = {
      {qkv4_w, wqkv4, C * C3},   {qkv8_w, wqkv8, C * C3},
      {qkv16_w, wqkv16, C * C3}, {sqkv_w, wsqkv, C * C3},
      {proj4_w, wproj4, C * C},  {proj8_w, wproj8, C * C},
      {proj16_w, wproj16, C * C},{tfc_w, wtfc, C * C},
      {sproj_w, wsproj, C * C},  {fc1_w, wfc1, C * C4},
      {fc2_w, wfc2, C4 * C}};
  for (auto& j : jobs)
    cvt_f32_bf16<<<(unsigned)((j.n + 255) / 256), 256, 0, stream>>>(j.s, j.d, j.n);

  // ---- temporal LN ----
  ln_xt<<<Mt, 256, 0, stream>>>(x, tn1_w, tn1_b, Abf);

  // ---- temporal attention at 3 scales ----
  struct Scale { int t; const bf16* wq; const bf16* wp; const float* pb; float* xo; }
  scales[] = {{4, wqkv4, wproj4, proj4_b, X4},
              {8, wqkv8, wproj8, proj8_b, X8},
              {16, wqkv16, wproj16, proj16_b, X16}};
  for (auto& sc : scales) {
    int Mrows = B_SZ * K_SZ * sc.t;
    gemm_bf16_wmma<0, 0, float><<<dim3(C3 / 128, Mrows / 64), 256, 0, stream>>>(
        Abf, C, sc.t, sc.wq, C3, Qf, C3, Mrows, (int)C3, (int)C, nullptr, nullptr, 0);
    temporal_attn<<<B_SZ * K_SZ, 256, 0, stream>>>(Qf, ctxbf, sc.t);
    gemm_bf16_wmma<0, 0, float><<<dim3(C / 128, Mrows / 64), 256, 0, stream>>>(
        ctxbf, C, 0, sc.wp, C, sc.xo, C, Mrows, (int)C, (int)C, sc.pb, nullptr, 0);
  }

  // ---- scale merging ----
  {
    size_t n8 = (size_t)B_SZ * K_SZ * 4 * C;
    merge_half<<<(unsigned)((n8 + 255) / 256), 256, 0, stream>>>(X8, X4, 8, B_SZ * K_SZ);
    size_t n16 = (size_t)B_SZ * K_SZ * 8 * C;
    merge_half<<<(unsigned)((n16 + 255) / 256), 256, 0, stream>>>(X16, X8, 16, B_SZ * K_SZ);
  }

  // ---- xt_full = res_temp @ tfc + tfc_b + x[:,1:,:] ----
  {
    size_t n = (size_t)Mt * C;
    cvt_f32_bf16<<<(unsigned)((n + 255) / 256), 256, 0, stream>>>(X16, ctxbf, n);
    gemm_bf16_wmma<0, 2, float><<<dim3(C / 128, Mt / 64), 256, 0, stream>>>(
        ctxbf, C, 0, wtfc, C, XTF, C, Mt, (int)C, (int)C, tfc_b, x, KT);
  }

  // ---- spatial path ----
  ln_xs<<<Ms, 256, 0, stream>>>(x, XTF, n1_w, n1_b, Abf);
  gemm_bf16_wmma<0, 0, float><<<dim3(C3 / 128, Ms / 64), 256, 0, stream>>>(
      Abf, C, 0, wsqkv, C3, Qf, C3, Ms, (int)C3, (int)C, nullptr, nullptr, 0);
  spatial_attn<<<dim3(NHEAD, B_SZ * T_SZ), 256, 0, stream>>>(Qf, ctxbf);
  gemm_bf16_wmma<0, 0, float><<<dim3(C / 128, Ms / 64), 256, 0, stream>>>(
      ctxbf, C, 0, wsproj, C, RSP, C, Ms, (int)C, (int)C, sproj_b, nullptr, 0);

  // ---- cls aggregation ----
  ln_rows<float><<<128, 256, 0, stream>>>(RSP, N_SP, ncls_w, ncls_b, CLSsp);
  cls_agg_kernel<<<B_SZ, 256, 0, stream>>>(CLSsp, CLSag);

  // ---- x_cat ----
  {
    size_t n = (size_t)Mc * C;
    build_xcat<<<(unsigned)((n + 255) / 256), 256, 0, stream>>>(x, XTF, RSP, CLSag, XCAT);
  }

  // ---- MLP: gelu(h2 @ fc1 + b1) @ fc2 + b2 + x_cat -> d_out ----
  ln_rows<bf16><<<Mc, 256, 0, stream>>>(XCAT, 1, n2_w, n2_b, Abf);
  bf16* act1 = (bf16*)Qf;  // reuse qkv scratch (154 MB as bf16)
  gemm_bf16_wmma<1, 0, bf16><<<dim3(C4 / 128, (Mc + 63) / 64), 256, 0, stream>>>(
      Abf, C, 0, wfc1, C4, act1, (int)C4, Mc, (int)C4, (int)C, fc1_b, nullptr, 0);
  gemm_bf16_wmma<0, 1, float><<<dim3(C / 128, (Mc + 63) / 64), 256, 0, stream>>>(
      act1, C4, 0, wfc2, C, out, (int)C, Mc, (int)C, (int)C4, fc2_b, XCAT, 0);
}
